// SelfAttn_81432579932367
// MI455X (gfx1250) — matrix operand order repaired and checked by reading, not compiled, not run on hardware
//
#include <hip/hip_runtime.h>
#include <hip/hip_bf16.h>

typedef __attribute__((ext_vector_type(16))) __bf16 v16bf;
typedef __attribute__((ext_vector_type(8)))  float  v8f;

#define NBATCH 8
#define S_LEN  2048
#define D_DIM  1024

// ---------------------------------------------------------------------------
// WMMA fragment loaders (contiguous k order, same for A and B, wave32)
// A: 16x32 bf16 (MxK). lane 0-15: row=lane, elems 0-15 = K 0..15
//                      lane 16-31: row=lane-16, elems 0-15 = K 16..31
// ---------------------------------------------------------------------------
__device__ __forceinline__ v16bf load_frag_a(const __bf16* __restrict__ p, int ld, int lane) {
    const int row  = lane & 15;
    const int half = lane >> 4;
    const __bf16* q = p + (size_t)row * ld + half * 16;
    union { v16bf v; uint4 u[2]; } r;
    r.u[0] = *reinterpret_cast<const uint4*>(q);        // K base .. +7
    r.u[1] = *reinterpret_cast<const uint4*>(q + 8);    // K base+8 .. +15
    return r.v;
}

// B: 32x16 bf16 (KxN). lanes 0-15 hold column n=lane, K 0..15; lanes 16-31: K 16..31.
// Source stored "row n contiguous over k" (i.e. W[n][k], K[s][d], Vt[d][s]).
__device__ __forceinline__ v16bf load_frag_b(const __bf16* __restrict__ p, int ld, int lane) {
    const int col  = lane & 15;
    const int koff = (lane >> 4) * 16;
    const __bf16* q = p + (size_t)col * ld + koff;
    union { v16bf v; uint4 u[2]; } r;
    r.u[0] = *reinterpret_cast<const uint4*>(q);        // K koff .. +7
    r.u[1] = *reinterpret_cast<const uint4*>(q + 8);    // K koff+8 .. +15
    return r.v;
}

__device__ __forceinline__ v8f wmma_bf16(v16bf a, v16bf b, v8f c) {
    return __builtin_amdgcn_wmma_f32_16x16x32_bf16(false, a, false, b, (short)0, c, false, false);
}

// ---------------------------------------------------------------------------
// Software-pipelined 32x64 wave-tile GEMM core.
// acc[i*4+j]: rows 16*i..16*i+15 of the wave tile, cols 16*j..16*j+15.
// Two disjoint fragment register sets (ping-pong) keep loads for the next
// k-step in flight while the WMMAs for the current k-step execute.
// klen must be a multiple of 64.
// ---------------------------------------------------------------------------
__device__ __forceinline__ void gemm_wave_32x64(
    const __bf16* __restrict__ aP, int lda,
    const __bf16* __restrict__ bP, int ldb,
    int klen, int lane, v8f acc[8])
{
    v16bf A0[2], B0[4], A1[2], B1[4];

    // prologue: k = 0
    A0[0] = load_frag_a(aP, lda, lane);
    A0[1] = load_frag_a(aP + (size_t)16 * lda, lda, lane);
#pragma unroll
    for (int j = 0; j < 4; ++j)
        B0[j] = load_frag_b(bP + (size_t)(16 * j) * ldb, ldb, lane);

    for (int k0 = 0; k0 < klen; k0 += 64) {
        // fetch k0+32 into set 1
        A1[0] = load_frag_a(aP + k0 + 32, lda, lane);
        A1[1] = load_frag_a(aP + (size_t)16 * lda + k0 + 32, lda, lane);
#pragma unroll
        for (int j = 0; j < 4; ++j)
            B1[j] = load_frag_b(bP + (size_t)(16 * j) * ldb + k0 + 32, ldb, lane);

        // compute on set 0 (k0)
#pragma unroll
        for (int i = 0; i < 2; ++i)
#pragma unroll
            for (int j = 0; j < 4; ++j)
                acc[i * 4 + j] = wmma_bf16(A0[i], B0[j], acc[i * 4 + j]);

        // fetch k0+64 into set 0
        if (k0 + 64 < klen) {
            A0[0] = load_frag_a(aP + k0 + 64, lda, lane);
            A0[1] = load_frag_a(aP + (size_t)16 * lda + k0 + 64, lda, lane);
#pragma unroll
            for (int j = 0; j < 4; ++j)
                B0[j] = load_frag_b(bP + (size_t)(16 * j) * ldb + k0 + 64, ldb, lane);
        }

        // compute on set 1 (k0+32)
#pragma unroll
        for (int i = 0; i < 2; ++i)
#pragma unroll
            for (int j = 0; j < 4; ++j)
                acc[i * 4 + j] = wmma_bf16(A1[i], B1[j], acc[i * 4 + j]);
    }
}

// ---------------------------------------------------------------------------
// 1) fp32 -> bf16 conversion (vectorized)
// ---------------------------------------------------------------------------
__global__ __launch_bounds__(256) void cvt_f32_to_bf16(const float* __restrict__ src,
                                                       __bf16* __restrict__ dst, int n) {
    int i = (blockIdx.x * 256 + threadIdx.x) * 4;
    if (i < n) {
        float4 v = *reinterpret_cast<const float4*>(src + i);
        union { __bf16 h[4]; uint2 u; } p;
        p.h[0] = (__bf16)v.x; p.h[1] = (__bf16)v.y;
        p.h[2] = (__bf16)v.z; p.h[3] = (__bf16)v.w;
        *reinterpret_cast<uint2*>(dst + i) = p.u;
    }
}

// ---------------------------------------------------------------------------
// 2) QKV projection: y = x @ W^T + b.  Block = 128x64 tile, 4 waves (wave: 32x64).
//    which = blockIdx.y>>4 selects Q/K/V.  Q gets *scale folded in.
//    V is stored TRANSPOSED: Vt[b][d][s], so PV GEMM B-frags are contiguous.
// ---------------------------------------------------------------------------
__global__ __launch_bounds__(128) void qkv_proj_kernel(
    const __bf16* __restrict__ xb,
    const __bf16* __restrict__ wqb, const __bf16* __restrict__ wkb, const __bf16* __restrict__ wvb,
    const float* __restrict__ bq, const float* __restrict__ bk, const float* __restrict__ bv,
    __bf16* __restrict__ Qb, __bf16* __restrict__ Kb, __bf16* __restrict__ Vt)
{
    const int lane  = threadIdx.x & 31;
    const int wave  = threadIdx.x >> 5;
    const int m0    = blockIdx.x * 128 + wave * 32;
    const int which = blockIdx.y >> 4;                 // 0=Q 1=K 2=V
    const int n0    = (blockIdx.y & 15) * 64;

    const __bf16* W    = (which == 0) ? wqb : (which == 1) ? wkb : wvb;
    const float*  bias = (which == 0) ? bq  : (which == 1) ? bk  : bv;

    v8f acc[8] = {};
    gemm_wave_32x64(xb + (size_t)m0 * D_DIM, D_DIM,
                    W + (size_t)n0 * D_DIM, D_DIM,
                    D_DIM, lane, acc);

    const int h = lane >> 4, nc = lane & 15;
#pragma unroll
    for (int i = 0; i < 2; ++i) {
#pragma unroll
        for (int j = 0; j < 4; ++j) {
            const int n = n0 + 16 * j + nc;
            const float bn = bias[n];
#pragma unroll
            for (int r = 0; r < 8; ++r) {
                const int m = m0 + 16 * i + r + 8 * h;
                float val = acc[i * 4 + j][r] + bn;
                if (which == 0) {
                    Qb[(size_t)m * D_DIM + n] = (__bf16)(val * 0.03125f);   // 1/sqrt(1024)
                } else if (which == 1) {
                    Kb[(size_t)m * D_DIM + n] = (__bf16)val;
                } else {
                    const int bb = m >> 11, s = m & 2047;
                    Vt[((size_t)(bb * D_DIM + n)) * S_LEN + s] = (__bf16)val;
                }
            }
        }
    }
}

// ---------------------------------------------------------------------------
// 3) Scores: Sr[b][q][k] = (Qs . K^T) + attn_bias  (scale already in Q)
//    Block computes one 128x64 tile; contraction over d=1024.
// ---------------------------------------------------------------------------
__global__ __launch_bounds__(128) void scores_kernel(
    const __bf16* __restrict__ Qb, const __bf16* __restrict__ Kb,
    const float* __restrict__ attn_bias, float* __restrict__ Sr)
{
    const int lane = threadIdx.x & 31;
    const int wave = threadIdx.x >> 5;
    const int kb0  = blockIdx.x * 64;
    const int m0   = blockIdx.y * 128 + wave * 32;
    const int b    = blockIdx.z;

    v8f acc[8] = {};
    gemm_wave_32x64(Qb + ((size_t)(b * S_LEN + m0)) * D_DIM, D_DIM,
                    Kb + ((size_t)(b * S_LEN + kb0)) * D_DIM, D_DIM,
                    D_DIM, lane, acc);

    const int h = lane >> 4, nc = lane & 15;
#pragma unroll
    for (int i = 0; i < 2; ++i) {
#pragma unroll
        for (int j = 0; j < 4; ++j) {
#pragma unroll
            for (int r = 0; r < 8; ++r) {
                const int m = m0 + 16 * i + r + 8 * h;
                const int k = kb0 + 16 * j + nc;
                const size_t idx = ((size_t)(b * S_LEN + m)) * S_LEN + k;
                Sr[idx] = acc[i * 4 + j][r] + attn_bias[idx];
            }
        }
    }
}

// ---------------------------------------------------------------------------
// 4) Row softmax: one wave32 per row. Loads full 2048-f32 row into registers
//    (float4), shfl_xor max/sum reductions, writes bf16 probs in place into
//    the front half of the row's 8KB slot (safe: row owned by one wave).
// ---------------------------------------------------------------------------
__global__ __launch_bounds__(256) void softmax_kernel(float* __restrict__ Sr)
{
    const int lane = threadIdx.x & 31;
    const int row  = blockIdx.x * 8 + (threadIdx.x >> 5);
    float* srow = Sr + (size_t)row * S_LEN;

    float vals[64];
    float m = -3.402823466e38f;
    const float4* s4 = reinterpret_cast<const float4*>(srow) + lane * 16;
#pragma unroll
    for (int i = 0; i < 16; ++i) {
        float4 v = s4[i];
        vals[4 * i + 0] = v.x; vals[4 * i + 1] = v.y;
        vals[4 * i + 2] = v.z; vals[4 * i + 3] = v.w;
        m = fmaxf(m, fmaxf(fmaxf(v.x, v.y), fmaxf(v.z, v.w)));
    }
#pragma unroll
    for (int off = 16; off > 0; off >>= 1) m = fmaxf(m, __shfl_xor(m, off));

    float s = 0.f;
#pragma unroll
    for (int j = 0; j < 64; ++j) { float e = __expf(vals[j] - m); vals[j] = e; s += e; }
#pragma unroll
    for (int off = 16; off > 0; off >>= 1) s += __shfl_xor(s, off);
    const float inv = 1.0f / s;

    __bf16* prow = reinterpret_cast<__bf16*>(srow);   // bf16 P, ld = 4096 elems/row slot
#pragma unroll
    for (int t = 0; t < 8; ++t) {
        union { __bf16 h[8]; uint4 u; } pk;
#pragma unroll
        for (int e = 0; e < 8; ++e) pk.h[e] = (__bf16)(vals[8 * t + e] * inv);
        *reinterpret_cast<uint4*>(prow + lane * 64 + t * 8) = pk.u;
    }
}

// ---------------------------------------------------------------------------
// 5) out = P @ V, using Vt[b][d][s] so B-frags are contiguous. ldA = 4096.
// ---------------------------------------------------------------------------
__global__ __launch_bounds__(128) void pv_kernel(
    const float* __restrict__ Sr, const __bf16* __restrict__ Vt, float* __restrict__ out)
{
    const int lane = threadIdx.x & 31;
    const int wave = threadIdx.x >> 5;
    const int m0 = blockIdx.x * 128 + wave * 32;    // global row (b*2048 + q)
    const int n0 = blockIdx.y * 64;                 // output d column
    const int b  = m0 >> 11;

    const __bf16* P = reinterpret_cast<const __bf16*>(Sr);

    v8f acc[8] = {};
    gemm_wave_32x64(P + (size_t)m0 * 4096, 4096,
                    Vt + ((size_t)(b * D_DIM + n0)) * S_LEN, S_LEN,
                    S_LEN, lane, acc);

    const int h = lane >> 4, nc = lane & 15;
#pragma unroll
    for (int i = 0; i < 2; ++i) {
#pragma unroll
        for (int j = 0; j < 4; ++j) {
#pragma unroll
            for (int r = 0; r < 8; ++r) {
                const int m = m0 + 16 * i + r + 8 * h;
                out[(size_t)m * D_DIM + n0 + 16 * j + nc] = acc[i * 4 + j][r];
            }
        }
    }
}

// ---------------------------------------------------------------------------
extern "C" void kernel_launch(void* const* d_in, const int* in_sizes, int n_in,
                              void* d_out, int out_size, void* d_ws, size_t ws_size,
                              hipStream_t stream) {
    (void)in_sizes; (void)n_in; (void)out_size; (void)ws_size;
    const float* x  = (const float*)d_in[0];
    const float* ab = (const float*)d_in[1];
    const float* Wq = (const float*)d_in[2];
    const float* bq = (const float*)d_in[3];
    const float* Wk = (const float*)d_in[4];
    const float* bk = (const float*)d_in[5];
    const float* Wv = (const float*)d_in[6];
    const float* bv = (const float*)d_in[7];
    float* out = (float*)d_out;

    // Workspace layout (~262 MB)
    char* w = (char*)d_ws;
    size_t off = 0;
    __bf16* xb  = (__bf16*)(w + off); off += (size_t)16384 * 1024 * 2;   // 32 MB
    __bf16* wqb = (__bf16*)(w + off); off += (size_t)1024 * 1024 * 2;    //  2 MB
    __bf16* wkb = (__bf16*)(w + off); off += (size_t)1024 * 1024 * 2;
    __bf16* wvb = (__bf16*)(w + off); off += (size_t)1024 * 1024 * 2;
    __bf16* Qb  = (__bf16*)(w + off); off += (size_t)16384 * 1024 * 2;
    __bf16* Kb  = (__bf16*)(w + off); off += (size_t)16384 * 1024 * 2;
    __bf16* Vt  = (__bf16*)(w + off); off += (size_t)16384 * 1024 * 2;
    float*  Sr  = (float*)(w + off);  // 16384 * 2048 * 4 = 128 MB (scores, then bf16 P in-place)

    cvt_f32_to_bf16<<<16384, 256, 0, stream>>>(x,  xb,  16384 * 1024);
    cvt_f32_to_bf16<<<1024,  256, 0, stream>>>(Wq, wqb, 1024 * 1024);
    cvt_f32_to_bf16<<<1024,  256, 0, stream>>>(Wk, wkb, 1024 * 1024);
    cvt_f32_to_bf16<<<1024,  256, 0, stream>>>(Wv, wvb, 1024 * 1024);

    qkv_proj_kernel<<<dim3(128, 48), 128, 0, stream>>>(xb, wqb, wkb, wvb, bq, bk, bv, Qb, Kb, Vt);
    scores_kernel  <<<dim3(32, 16, 8), 128, 0, stream>>>(Qb, Kb, ab, Sr);
    softmax_kernel <<<2048, 256, 0, stream>>>(Sr);
    pv_kernel      <<<dim3(128, 16), 128, 0, stream>>>(Sr, Vt, out);
}
